// RNN_model_59322088293066
// MI455X (gfx1250) — matrix operand order repaired and checked by reading, not compiled, not run on hardware
//
#include <hip/hip_runtime.h>
#include <hip/hip_bf16.h>

// ---------------------------------------------------------------------------
// LSTM, H=2048, 512 sequential steps, batch 1.
//   step 0   : gates = x0 @ W_ih.T + (b_ih + b_hh)
//   steps>=1 : gates = h  @ (W_ih + W_hh).T + (b_ih + b_hh)   (since x==h)
// W_sum folded to bf16 (32 MB -> L2-resident), pre-swizzled into WMMA
// B-fragment order. Per step: one kernel, 128 WGs x 8 waves; each wave runs
// v_wmma_f32_16x16x32_bf16 over its K-range with TWO accumulator chains and
// a 4-chunk-unrolled body (loads grouped ahead of WMMAs for MLP overlap).
// Fixed-order LDS combine + fp32 pointwise; h ping-pongs (bf16).
// ---------------------------------------------------------------------------

typedef __attribute__((ext_vector_type(16))) __bf16 v16bf;
typedef __attribute__((ext_vector_type(8)))  float  v8f;

union Frag {
    uint4 u[2];
    v16bf v;
};

#define H 2048
#define G4 8192  // 4*H

static __device__ __forceinline__ unsigned short f2bf(float f) {
    unsigned int u = __float_as_uint(f);
    // round-to-nearest-even bf16 truncation
    u = (u + 0x7FFFu + ((u >> 16) & 1u)) >> 16;
    return (unsigned short)u;
}

static __device__ __forceinline__ float sigf(float x) {
    return 1.0f / (1.0f + __expf(-x));
}

// A (16x32 bf16): all rows = h chunk. lanes 0-15: K 0..15,
// lanes 16-31: K 16..31 -> two b128 loads (L0-hot, broadcast).
static __device__ __forceinline__ void loadA(Frag& f, const uint4* __restrict__ hv,
                                             int ch, int sel) {
    int hbase = ch * 4 + sel * 2;
    f.u[0] = hv[hbase];
    f.u[1] = hv[hbase + 1];
}

// B (32x16 bf16): pre-swizzled tile, lane = 32 contiguous bytes.
static __device__ __forceinline__ void loadB(Frag& f, const uint4* __restrict__ Wv,
                                             int ch, int lane2) {
    const uint4* tp = Wv + ch * 64 + lane2;
    f.u[0] = tp[0];
    f.u[1] = tp[1];
}

#define WMMA_BF16(A, B, C) \
    __builtin_amdgcn_wmma_f32_16x16x32_bf16(false, (A).v, false, (B).v, (short)0, (C), false, false)

// ---- bias fold: b = b_ih + b_hh (8192) -------------------------------------
__global__ void __launch_bounds__(256) k_bias(const float* __restrict__ b_ih,
                                              const float* __restrict__ b_hh,
                                              float* __restrict__ bias) {
    int i = blockIdx.x * 256 + threadIdx.x;
    bias[i] = b_ih[i] + b_hh[i];
}

// ---- weight fold + swizzle into WMMA B-fragment layout ---------------------
// 512 panels (16 rows) x 64 k-chunks (32 cols). Tile = 512 bf16 = 32 lanes x
// 16 values; lane l holds row (l&15), cols chunk*32 + (l>>4)*16 .. +15
// (K packed 2-per-dword, low = even K).
__global__ void __launch_bounds__(256) k_convw(const float* __restrict__ Wih,
                                               const float* __restrict__ Whh,
                                               unsigned short* __restrict__ Wsw) {
    int t    = blockIdx.x * 256 + threadIdx.x;  // 0 .. 1048575
    int tile = t >> 5;
    int lane = t & 31;
    int panel = tile >> 6;
    int chunk = tile & 63;
    int row = panel * 16 + (lane & 15);
    int col = chunk * 32 + (lane >> 4) * 16;
    size_t src = (size_t)row * H + col;

    unsigned int pk[8];
#pragma unroll
    for (int e = 0; e < 8; ++e) {
        float a0 = Wih[src + 2 * e]     + Whh[src + 2 * e];
        float a1 = Wih[src + 2 * e + 1] + Whh[src + 2 * e + 1];
        pk[e] = (unsigned int)f2bf(a0) | ((unsigned int)f2bf(a1) << 16);
    }
    uint4* dst = (uint4*)(Wsw + (size_t)tile * 512 + lane * 16);
    dst[0] = make_uint4(pk[0], pk[1], pk[2], pk[3]);
    dst[1] = make_uint4(pk[4], pk[5], pk[6], pk[7]);
}

// ---- pointwise LSTM update (wave 0, lanes 0..15 of each WG) ----------------
static __device__ __forceinline__ void lstm_pointwise(
    const float* lds, const float* __restrict__ bias, int jb, int lane,
    float c_old, float* __restrict__ c, float* __restrict__ h_f32,
    unsigned short* __restrict__ h_bf) {
    int j = jb * 16 + lane;
    float gi = lds[0 * 16 + lane] + lds[4 * 16 + lane] + bias[0 * H + j];
    float gf = lds[1 * 16 + lane] + lds[5 * 16 + lane] + bias[1 * H + j];
    float gg = lds[2 * 16 + lane] + lds[6 * 16 + lane] + bias[2 * H + j];
    float go = lds[3 * 16 + lane] + lds[7 * 16 + lane] + bias[3 * H + j];
    float i_ = sigf(gi);
    float f_ = sigf(gf);
    float g_ = tanhf(gg);
    float o_ = sigf(go);
    float cn = f_ * c_old + i_ * g_;
    float hn = o_ * tanhf(cn);
    c[j]     = cn;
    h_f32[j] = hn;
    h_bf[j]  = f2bf(hn);
}

// ---- step 0: fp32 GEMV over W_ih with x0, h=0, c=0 -------------------------
__global__ void __launch_bounds__(256) k_step0(const float* __restrict__ Wih,
                                               const float* __restrict__ x0,
                                               const float* __restrict__ bias,
                                               float* __restrict__ c,
                                               float* __restrict__ h_f32,
                                               unsigned short* __restrict__ h_bf) {
    __shared__ float lds[128];
    int jb   = blockIdx.x;
    int w    = threadIdx.x >> 5;
    int lane = threadIdx.x & 31;
    int gate  = w & 3;
    int khalf = w >> 2;  // 0 or 1, K range of 1024

#pragma unroll 1
    for (int r = 0; r < 16; ++r) {
        const float* row = Wih + (size_t)(gate * H + jb * 16 + r) * H + khalf * 1024;
        const float* xv  = x0 + khalf * 1024;
        float s = 0.f;
        for (int k = lane; k < 1024; k += 32) s += row[k] * xv[k];
#pragma unroll
        for (int off = 16; off > 0; off >>= 1) s += __shfl_down(s, off, 32);
        if (lane == 0) lds[w * 16 + r] = s;
    }
    __syncthreads();
    if (w == 0 && lane < 16)
        lstm_pointwise(lds, bias, jb, lane, /*c_old=*/0.0f, c, h_f32, h_bf);
}

// ---- steps 1..511: bf16 WMMA GEMV over pre-swizzled W_sum ------------------
__global__ void __launch_bounds__(256) k_step(const unsigned short* __restrict__ Wsw,
                                              const float* __restrict__ bias,
                                              const unsigned short* __restrict__ h_in,
                                              unsigned short* __restrict__ h_out,
                                              float* __restrict__ c,
                                              float* __restrict__ h_f32) {
    __shared__ float lds[128];
    int jb   = blockIdx.x;                 // output block: h[jb*16 .. +15]
    int w    = threadIdx.x >> 5;
    int lane = threadIdx.x & 31;
    int gate  = w & 3;
    int khalf = w >> 2;

    int panel = gate * 128 + jb;           // 16-row panel of W_sum (64 KB)
    const uint4* Wv = (const uint4*)Wsw + (size_t)panel * 4096;
    const uint4* hv = (const uint4*)h_in;
    int sel   = lane >> 4;
    int lane2 = lane * 2;

    // Two independent accumulator chains (break WMMA->WMMA RAW), 4 chunks per
    // iteration with loads grouped ahead of the WMMAs so the next group's 16
    // b128 loads stream from L2 while the XDL pipe runs the current 4 WMMAs.
    v8f acc0 = {}, acc1 = {};
    int c0 = khalf * 32;
#pragma unroll 1
    for (int ch = c0; ch < c0 + 32; ch += 4) {
        Frag A0, B0, A1, B1, A2, B2, A3, B3;
        loadA(A0, hv, ch + 0, sel);  loadB(B0, Wv, ch + 0, lane2);
        loadA(A1, hv, ch + 1, sel);  loadB(B1, Wv, ch + 1, lane2);
        loadA(A2, hv, ch + 2, sel);  loadB(B2, Wv, ch + 2, lane2);
        loadA(A3, hv, ch + 3, sel);  loadB(B3, Wv, ch + 3, lane2);
        acc0 = WMMA_BF16(A0, B0, acc0);
        acc1 = WMMA_BF16(A1, B1, acc1);
        acc0 = WMMA_BF16(A2, B2, acc0);
        acc1 = WMMA_BF16(A3, B3, acc1);
    }
    // D row 0 (VGPR0, lanes 0..15) holds the 16 partial dot products.
    if (lane < 16) lds[w * 16 + lane] = acc0[0] + acc1[0];
    __syncthreads();
    if (w == 0 && lane < 16)
        lstm_pointwise(lds, bias, jb, lane, c[jb * 16 + lane], c, h_f32, h_out);
}

// ---- logits: 11 x 2048 GEMV ------------------------------------------------
__global__ void __launch_bounds__(256) k_logits(const float* __restrict__ h,
                                                const float* __restrict__ W_out,
                                                const float* __restrict__ b_out,
                                                float* __restrict__ logits) {
    __shared__ float red[256];
    int r = blockIdx.x, t = threadIdx.x;
    float s = 0.f;
    for (int k = t; k < H; k += 256) s += h[k] * W_out[(size_t)r * H + k];
    red[t] = s;
    __syncthreads();
    for (int o = 128; o > 0; o >>= 1) {
        if (t < o) red[t] += red[t + o];
        __syncthreads();
    }
    if (t == 0) logits[r] = red[0] + b_out[r];
}

// ---- softmax + pack outputs (probs[11], h[2048], c[2048]) ------------------
__global__ void __launch_bounds__(256) k_final(const float* __restrict__ logits,
                                               const float* __restrict__ h,
                                               const float* __restrict__ c,
                                               float* __restrict__ out) {
    int t = threadIdx.x;
    if (t == 0) {
        float m = -3.0e38f;
        for (int i = 0; i < 11; ++i) m = fmaxf(m, logits[i]);
        float e[11], s = 0.f;
        for (int i = 0; i < 11; ++i) { e[i] = __expf(logits[i] - m); s += e[i]; }
        float inv = 1.0f / s;
        for (int i = 0; i < 11; ++i) out[i] = e[i] * inv;
    }
    for (int k = t; k < H; k += 256) {
        out[11 + k]     = h[k];
        out[11 + H + k] = c[k];
    }
}

extern "C" void kernel_launch(void* const* d_in, const int* in_sizes, int n_in,
                              void* d_out, int out_size, void* d_ws, size_t ws_size,
                              hipStream_t stream) {
    const float* inputs = (const float*)d_in[0];  // (512, 2048); use row 0
    const float* W_ih   = (const float*)d_in[1];  // (8192, 2048)
    const float* W_hh   = (const float*)d_in[2];  // (8192, 2048)
    const float* b_ih   = (const float*)d_in[3];  // (8192,)
    const float* b_hh   = (const float*)d_in[4];  // (8192,)
    const float* W_out  = (const float*)d_in[5];  // (11, 2048)
    const float* b_out  = (const float*)d_in[6];  // (11,)
    // d_in[7] = steps (fixed 512 by setup)
    (void)in_sizes; (void)n_in; (void)out_size;

    // workspace layout
    char* ws = (char*)d_ws;
    unsigned short* Wsw  = (unsigned short*)ws;                       // 32 MB
    float*          bias = (float*)(ws + 33554432);                   // 32 KB
    unsigned short* hb0  = (unsigned short*)(ws + 33554432 + 32768);  // 4 KB
    unsigned short* hb1  = hb0 + H;                                   // 4 KB
    float*          cbuf = (float*)(ws + 33554432 + 32768 + 8192);    // 8 KB
    float*          hf   = cbuf + H;                                  // 8 KB
    float*          logs = hf + H;                                    // 44 B
    (void)ws_size;

    k_bias <<<32,   256, 0, stream>>>(b_ih, b_hh, bias);
    k_convw<<<4096, 256, 0, stream>>>(W_ih, W_hh, Wsw);
    k_step0<<<128,  256, 0, stream>>>(W_ih, inputs, bias, cbuf, hf, hb0);

    for (int s = 1; s < 512; ++s) {
        const unsigned short* hin  = (s & 1) ? hb0 : hb1;
        unsigned short*       hout = (s & 1) ? hb1 : hb0;
        k_step<<<128, 256, 0, stream>>>(Wsw, bias, hin, hout, cbuf, hf);
    }

    k_logits<<<11, 256, 0, stream>>>(hf, W_out, b_out, logs);
    k_final <<<1,  256, 0, stream>>>(logs, hf, cbuf, (float*)d_out);
}
